// MaskedAutoencoderBNTF_76914274337443
// MI455X (gfx1250) — matrix operand order repaired and checked by reading, not compiled, not run on hardware
//
#include <hip/hip_runtime.h>

// ---------------- problem constants ----------------
#define B_SZ   256
#define NN     100           // nodes
#define DD     100           // feature dim
#define DPAD   128           // padded feature dim (f16 activation stride)
#define BN     (B_SZ * NN)   // 25600 rows
#define FFD    2048
#define PHD    1024
#define KNET   7
#define NHEADC 4
#define HDIMC  25
#define CHUNK  3200          // FF row chunk (keeps hidden buf at 13MB f16, inside L2)

typedef __attribute__((ext_vector_type(16))) _Float16 v16h;
typedef __attribute__((ext_vector_type(8)))  _Float16 v8h;
typedef __attribute__((ext_vector_type(8)))  float    v8f;

// subnet membership (0-based node -> cluster id), counts = {17,14,15,12,5,13,24}
__device__ __constant__ unsigned char NODE_K[100] = {
  0,0,0,0,0,0,0,0,0,            // 0-8
  1,1,1,1,1,1,                  // 9-14
  2,2,2,2,2,2,2,2,              // 15-22
  3,3,3,3,3,3,3,                // 23-29
  4,4,4,                        // 30-32
  5,5,5,5,                      // 33-36
  6,6,6,6,6,6,6,6,6,6,6,6,6,    // 37-49
  0,0,0,0,0,0,0,0,              // 50-57
  1,1,1,1,1,1,1,1,              // 58-65
  2,2,2,2,2,2,2,                // 66-72
  3,3,3,3,3,                    // 73-77
  4,4,                          // 78-79
  5,5,5,5,5,5,5,5,5,            // 80-88
  6,6,6,6,6,6,6,6,6,6,6         // 89-99
};
__device__ __constant__ float KCNT[7] = {17.f,14.f,15.f,12.f,5.f,13.f,24.f};

__device__ __forceinline__ double rho_of(const int* ep) {
  double e = (double)ep[0];
  double u = 1.0 - e / 600.0;
  return 0.8 + 0.2 * exp(-5.0 * u * u);
}

#define SHUF16(lo, hi) __builtin_shufflevector(lo, hi, 0,1,2,3,4,5,6,7,8,9,10,11,12,13,14,15)

// ---------------- WMMA GEMM on pre-padded f16 operands ----------------
// Ah: [M][Kpad] f16 (zero-padded), Bt: [Npad][Kpad] f16 (transposed, zero-padded).
// One wave computes a 32x32 register tile: per 32-K panel 8 unconditional b128
// loads + 4 v_wmma_f32_16x16x32_f16. No LDS, no barriers, no edge paths.
// Outputs: C f32 [M][N] (optional, col-guarded) and/or Ch f16 [M][Npad] (optional).
// act: 0 none, 1 relu, 2 leaky_relu(0.01). Requires M%32==0, Npad%32==0, Kpad%32==0.
__global__ __launch_bounds__(256)
void gemm_kernel(const _Float16* __restrict__ Ah, const _Float16* __restrict__ Bt,
                 const float* __restrict__ bias, float* __restrict__ C,
                 _Float16* __restrict__ Ch,
                 int M, int N, int Npad, int Kpad, int act)
{
  const int wave = threadIdx.x >> 5;
  const int lane = threadIdx.x & 31;
  const int tNp  = Npad >> 5;
  const int tiles = (M >> 5) * tNp;
  const int tile = blockIdx.x * 8 + wave;
  if (tile >= tiles) return;                 // uniform per wave: EXEC all-1 for WMMA
  const int m0 = (tile / tNp) * 32;
  const int n0 = (tile % tNp) * 32;
  const int hsel = lane >> 4, l15 = lane & 15;

  // A frag (16x32): e<8 -> k=hsel*8+e ; e>=8 -> k=16+hsel*8+(e-8)
  const _Float16* a0 = Ah + (size_t)(m0 + l15) * (size_t)Kpad + hsel * 8;
  const _Float16* a1 = a0 + (size_t)16 * (size_t)Kpad;
  // B frag (32x16): lane = column, e<8 -> k=hsel*8+e ; e>=8 -> k=16+hsel*8+(e-8)
  const _Float16* b0 = Bt + (size_t)(n0 + l15) * (size_t)Kpad + hsel * 8;
  const _Float16* b1 = b0 + (size_t)16 * (size_t)Kpad;

  v8f acc00 = {0.f,0.f,0.f,0.f,0.f,0.f,0.f,0.f};
  v8f acc01 = {0.f,0.f,0.f,0.f,0.f,0.f,0.f,0.f};
  v8f acc10 = {0.f,0.f,0.f,0.f,0.f,0.f,0.f,0.f};
  v8f acc11 = {0.f,0.f,0.f,0.f,0.f,0.f,0.f,0.f};

  for (int kb = 0; kb < Kpad; kb += 32) {
    v8h a0l = *(const v8h*)(a0 + kb);
    v8h a0h = *(const v8h*)(a0 + kb + 16);
    v8h a1l = *(const v8h*)(a1 + kb);
    v8h a1h = *(const v8h*)(a1 + kb + 16);
    v8h b0l = *(const v8h*)(b0 + kb);
    v8h b0h = *(const v8h*)(b0 + kb + 16);
    v8h b1l = *(const v8h*)(b1 + kb);
    v8h b1h = *(const v8h*)(b1 + kb + 16);

    v16h af0 = SHUF16(a0l, a0h);
    v16h af1 = SHUF16(a1l, a1h);
    v16h bf0 = SHUF16(b0l, b0h);
    v16h bf1 = SHUF16(b1l, b1h);

    acc00 = __builtin_amdgcn_wmma_f32_16x16x32_f16(false, af0, false, bf0, (short)0, acc00, false, false);
    acc01 = __builtin_amdgcn_wmma_f32_16x16x32_f16(false, af0, false, bf1, (short)0, acc01, false, false);
    acc10 = __builtin_amdgcn_wmma_f32_16x16x32_f16(false, af1, false, bf0, (short)0, acc10, false, false);
    acc11 = __builtin_amdgcn_wmma_f32_16x16x32_f16(false, af1, false, bf1, (short)0, acc11, false, false);
  }

  // D layout: lane = column, VGPR r -> row r + 8*hsel
  #pragma unroll
  for (int i = 0; i < 2; ++i) {
    #pragma unroll
    for (int j = 0; j < 2; ++j) {
      v8f acc = (i == 0) ? (j == 0 ? acc00 : acc01) : (j == 0 ? acc10 : acc11);
      const int col = n0 + j * 16 + l15;
      const float bv = (col < N) ? bias[col] : 0.0f;
      #pragma unroll
      for (int r = 0; r < 8; ++r) {
        int m = m0 + i * 16 + r + 8 * hsel;
        float v = acc[r] + bv;
        if (act == 1)      v = v > 0.f ? v : 0.f;
        else if (act == 2) v = v > 0.f ? v : 0.01f * v;
        if (C && col < N) C[(size_t)m * (size_t)N + col] = v;
        if (Ch)           Ch[(size_t)m * (size_t)Npad + col] = (_Float16)v;
      }
    }
  }
}

// ---------------- weight pack: Bt[n][k] = W[k][n], zero-padded, f16 ----------------
__global__ void wt_pack_kernel(const float* __restrict__ W, _Float16* __restrict__ Bt,
                               int K, int N, int Kpad, int Npad)
{
  int i = blockIdx.x * 256 + threadIdx.x;
  if (i >= Npad * Kpad) return;
  int n = i / Kpad, k = i % Kpad;
  float v = (n < N && k < K) ? W[(size_t)k * (size_t)N + n] : 0.0f;
  Bt[i] = (_Float16)v;
}

// ---------------- activation pack: dst[m][Kpad] f16 from src[m][K] f32 ----------------
__global__ void a_pack_kernel(const float* __restrict__ src, _Float16* __restrict__ dst,
                              int M, int K, int Kpad)
{
  size_t i = (size_t)blockIdx.x * 256 + threadIdx.x;
  if (i >= (size_t)M * Kpad) return;
  int k = (int)(i % Kpad);
  int m = (int)(i / Kpad);
  dst[i] = (_Float16)((k < K) ? src[(size_t)m * K + k] : 0.0f);
}

__global__ void zero_h_kernel(_Float16* __restrict__ p, size_t count)
{
  size_t i = (size_t)blockIdx.x * 256 + threadIdx.x;
  if (i < count) p[i] = (_Float16)0.0f;
}

// ---------------- attention: one block per (batch, head); f16 padded output ----------------
__global__ __launch_bounds__(128)
void attn_kernel(const float* __restrict__ qkv, _Float16* __restrict__ oh)
{
  const int b = blockIdx.x >> 2;
  const int h = blockIdx.x & 3;
  const int tid = threadIdx.x;
  __shared__ float kb[NN][HDIMC];
  __shared__ float vb[NN][HDIMC];
  __shared__ float sc[NN][NN + 1];

  for (int i = tid; i < NN * HDIMC; i += 128) {
    int n = i / HDIMC, d = i % HDIMC;
    size_t base = ((size_t)(b * NN + n)) * 300;
    kb[n][d] = qkv[base + 100 + h * HDIMC + d];
    vb[n][d] = qkv[base + 200 + h * HDIMC + d];
  }
  __syncthreads();

  for (int t = tid; t < NN; t += 128) {
    float q[HDIMC];
    size_t qbase = ((size_t)(b * NN + t)) * 300 + h * HDIMC;
    #pragma unroll
    for (int d = 0; d < HDIMC; ++d) q[d] = qkv[qbase + d];

    float mx = -1e30f;
    for (int j = 0; j < NN; ++j) {
      float s = 0.f;
      #pragma unroll
      for (int d = 0; d < HDIMC; ++d) s += q[d] * kb[j][d];
      s *= 0.2f;                       // 1/sqrt(25)
      sc[t][j] = s;
      mx = fmaxf(mx, s);
    }
    float sum = 0.f;
    for (int j = 0; j < NN; ++j) {
      float e = expf(sc[t][j] - mx);
      sc[t][j] = e;
      sum += e;
    }
    float inv = 1.f / sum;
    size_t obase = ((size_t)(b * NN + t)) * DPAD + h * HDIMC;
    for (int d = 0; d < HDIMC; ++d) {
      float a = 0.f;
      for (int j = 0; j < NN; ++j) a += sc[t][j] * vb[j][d];
      oh[obase + d] = (_Float16)(a * inv);
    }
  }
}

// ---------------- residual add + layernorm; dual f32 + padded-f16 output ----------------
__global__ __launch_bounds__(256)
void add_ln_kernel(const float* __restrict__ x, const float* __restrict__ r,
                   const float* __restrict__ g, const float* __restrict__ bb,
                   float* __restrict__ y, _Float16* __restrict__ yh, int rows)
{
  const int wave = threadIdx.x >> 5, lane = threadIdx.x & 31;
  const int row = blockIdx.x * 8 + wave;
  if (row >= rows) return;
  const float* xp = x + (size_t)row * DD;
  const float* rp = r + (size_t)row * DD;
  float vals[4]; float s = 0.f, s2 = 0.f;
  #pragma unroll
  for (int i = 0; i < 4; ++i) {
    int d = lane + i * 32;
    float v = (d < DD) ? (xp[d] + rp[d]) : 0.f;
    vals[i] = v; s += v; s2 += v * v;
  }
  #pragma unroll
  for (int off = 16; off; off >>= 1) {
    s  += __shfl_xor(s,  off, 32);
    s2 += __shfl_xor(s2, off, 32);
  }
  float m   = s  * 0.01f;
  float var = s2 * 0.01f - m * m;
  float inv = rsqrtf(var + 1e-5f);
  #pragma unroll
  for (int i = 0; i < 4; ++i) {
    int d = lane + i * 32;                 // 0..127
    float outv = 0.f;
    if (d < DD) {
      outv = (vals[i] - m) * inv * g[d] + bb[d];
      y[(size_t)row * DD + d] = outv;
    }
    yh[(size_t)row * DPAD + d] = (_Float16)outv;   // zero in pad columns
  }
}

// ---------------- sinusoidal position table ----------------
__global__ void pos_kernel(float* __restrict__ pos)
{
  int i = blockIdx.x * 256 + threadIdx.x;
  if (i >= NN * DD) return;
  int p = i / DD, d = i % DD;
  double ang = (double)p / pow(10000.0, (double)(2 * (d / 2)) / 100.0);
  pos[i] = (float)((d & 1) ? cos(ang) : sin(ang));
}

// ---------------- subnet means: sub[b][k][d] ----------------
__global__ __launch_bounds__(256)
void subnet_kernel(const float* __restrict__ feat, float* __restrict__ sub)
{
  const int b = blockIdx.x;
  for (int i = threadIdx.x; i < KNET * DD; i += 256) {
    int k = i / DD, d = i % DD;
    float s = 0.f;
    for (int n = 0; n < NN; ++n)
      if (NODE_K[n] == k) s += feat[((size_t)(b * NN + n)) * DD + d];
    sub[(size_t)b * (KNET * DD) + i] = s / (KCNT[k] + 1e-6f);
  }
}

// ---------------- cosine cost: cost[b][n][k] ----------------
__global__ __launch_bounds__(256)
void cost_kernel(const float* __restrict__ feat, const float* __restrict__ sub,
                 float* __restrict__ cost)
{
  const int b = blockIdx.x; const int tid = threadIdx.x;
  __shared__ float s_sub[KNET][DD];
  __shared__ float s_sn[KNET];
  __shared__ float s_fn[NN];
  for (int i = tid; i < KNET * DD; i += 256)
    s_sub[i / DD][i % DD] = sub[(size_t)b * (KNET * DD) + i];
  __syncthreads();
  if (tid < NN) {
    const float* f = feat + ((size_t)(b * NN + tid)) * DD;
    float s = 0.f;
    for (int d = 0; d < DD; ++d) s += f[d] * f[d];
    s_fn[tid] = fmaxf(sqrtf(s), 1e-12f);
  } else if (tid < NN + KNET) {
    int k = tid - NN; float s = 0.f;
    for (int d = 0; d < DD; ++d) s += s_sub[k][d] * s_sub[k][d];
    s_sn[k] = fmaxf(sqrtf(s), 1e-12f);
  }
  __syncthreads();
  for (int i = tid; i < NN * KNET; i += 256) {
    int n = i / KNET, k = i % KNET;
    const float* f = feat + ((size_t)(b * NN + n)) * DD;
    float s = 0.f;
    for (int d = 0; d < DD; ++d) s += f[d] * s_sub[k][d];
    cost[(size_t)b * (NN * KNET) + i] = s / (s_fn[n] * s_sn[k]);
  }
}

// ---------------- FP64 Sinkhorn, one block per batch element ----------------
__global__ __launch_bounds__(128)
void sinkhorn_kernel(const float* __restrict__ cost, const int* __restrict__ ep,
                     float* __restrict__ Qout)
{
  const int b = blockIdx.x; const int tid = threadIdx.x;
  __shared__ double Qs[NN][KNET + 1];
  __shared__ double av[NN];
  __shared__ double bv[KNET + 1], bnw[KNET + 1];
  __shared__ int s_done;
  const double rho = rho_of(ep);
  const double FI  = 1.0 / 1.1;               // GAMMA/(GAMMA+EPS)

  for (int n = tid; n < NN; n += 128) {
    double c[KNET]; double mx = -1e300;
    for (int k = 0; k < KNET; ++k) {
      c[k] = (double)cost[(size_t)b * (NN * KNET) + n * KNET + k];
      mx = fmax(mx, c[k]);
    }
    double se = 0.0;
    for (int k = 0; k < KNET; ++k) se += exp(c[k] - mx);
    double lse = mx + log(se);
    for (int k = 0; k < KNET; ++k) Qs[n][k] = exp((c[k] - lse) * 10.0); // exp(-P/eps)
    Qs[n][KNET] = 1.0;
  }
  if (tid < KNET + 1) bv[tid] = 1.0 / 8.0;
  if (tid == 0) s_done = 0;
  __syncthreads();

  for (int it = 0; it < 50; ++it) {
    for (int n = tid; n < NN; n += 128) {
      double s = 0.0;
      for (int k = 0; k <= KNET; ++k) s += Qs[n][k] * bv[k];
      av[n] = 0.01 / s;                        // Pa = 1/100
    }
    __syncthreads();
    if (tid <= KNET) {
      double s = 0.0;
      for (int n = 0; n < NN; ++n) s += Qs[n][tid] * av[n];
      double Pbk = (tid < KNET) ? rho * (double)KCNT[tid] / 100.0 : (1.0 - rho);
      double v = Pbk / s;
      if (tid < KNET) v = pow(v, FI);
      bnw[tid] = v;
    }
    __syncthreads();
    if (tid == 0) {
      double e2 = 0.0;
      for (int k = 0; k <= KNET; ++k) { double d = bnw[k] - bv[k]; e2 += d * d; }
      for (int k = 0; k <= KNET; ++k) bv[k] = bnw[k];
      s_done = (e2 <= 1e-12);                  // ||b2-b|| <= 1e-6
    }
    __syncthreads();
    if (s_done) break;
  }

  for (int i = tid; i < NN * KNET; i += 128) {
    int n = i / KNET, k = i % KNET;
    Qout[(size_t)b * (NN * KNET) + i] = (float)(100.0 * av[n] * Qs[n][k] * bv[k]);
  }
}

// ---------------- quantile threshold + stable-rank mask ----------------
__global__ __launch_bounds__(128)
void mask_kernel(const float* __restrict__ Qp, const float* __restrict__ randK,
                 const int* __restrict__ ep, float* __restrict__ mask)
{
  const int b = blockIdx.x; const int tid = threadIdx.x;
  __shared__ float qual[NN], sorted[NN], noise[NN];
  __shared__ float thr;
  const float rho = (float)rho_of(ep);

  for (int n = tid; n < NN; n += 128) {
    float s = 0.f;
    for (int k = 0; k < KNET; ++k) s += Qp[(size_t)b * (NN * KNET) + n * KNET + k];
    qual[n] = s;
  }
  __syncthreads();
  for (int n = tid; n < NN; n += 128) {
    int r = 0; float qn = qual[n];
    for (int j = 0; j < NN; ++j)
      r += (qual[j] < qn) || (qual[j] == qn && j < n);
    sorted[r] = qn;
  }
  __syncthreads();
  if (tid == 0) {
    float t = 99.0f * (1.0f - rho);            // linear-interp quantile position
    int i0 = (int)floorf(t); if (i0 > 98) i0 = 98; if (i0 < 0) i0 = 0;
    float fr = t - (float)i0;
    thr = sorted[i0] + (sorted[i0 + 1] - sorted[i0]) * fr;
  }
  __syncthreads();
  for (int n = tid; n < NN; n += 128) {
    float v = 0.f;
    if (!(qual[n] < thr)) {
      for (int k = 0; k < KNET; ++k)
        v += Qp[(size_t)b * (NN * KNET) + n * KNET + k] * randK[b * KNET + k];
    }
    noise[n] = v;
  }
  __syncthreads();
  for (int n = tid; n < NN; n += 128) {
    int r = 0; float nv = noise[n];
    for (int j = 0; j < NN; ++j)
      r += (noise[j] < nv) || (noise[j] == nv && j < n);
    mask[b * NN + n] = (r < 90) ? 0.f : 1.f;   // LEN_KEEP = 90
  }
}

// ---------------- build masked input: f32 + padded f16 ----------------
__global__ void build_x_kernel(const float* __restrict__ imgs, const float* __restrict__ mtok,
                               const float* __restrict__ mask, const float* __restrict__ pos,
                               float* __restrict__ x, _Float16* __restrict__ xh)
{
  size_t i = (size_t)blockIdx.x * 256 + threadIdx.x;
  if (i >= (size_t)BN * DPAD) return;
  int d  = (int)(i % DPAD);
  int bn = (int)(i / DPAD);
  if (d < DD) {
    int n = bn % NN;
    float v = (mask[bn] == 0.f) ? imgs[(size_t)bn * DD + d] : mtok[d];
    v += pos[n * DD + d];
    x[(size_t)bn * DD + d] = v;
    xh[i] = (_Float16)v;
  } else {
    xh[i] = (_Float16)0.0f;
  }
}

// ---------------- deterministic masked-MSE reduction ----------------
__global__ __launch_bounds__(128)
void loss_partial_kernel(const float* __restrict__ pred, const float* __restrict__ imgs,
                         const float* __restrict__ mask, float* __restrict__ part)
{
  const int b = blockIdx.x; const int tid = threadIdx.x;
  __shared__ float sn[128], sd[128];
  float num = 0.f, den = 0.f;
  for (int n = tid; n < NN; n += 128) {
    const float* pp = pred + ((size_t)(b * NN + n)) * DD;
    const float* ip = imgs + ((size_t)(b * NN + n)) * DD;
    float s = 0.f;
    for (int d = 0; d < DD; ++d) { float df = pp[d] - ip[d]; s += df * df; }
    float m = mask[b * NN + n];
    num += s * 0.01f * m;
    den += m;
  }
  sn[tid] = num; sd[tid] = den; __syncthreads();
  for (int off = 64; off; off >>= 1) {
    if (tid < off) { sn[tid] += sn[tid + off]; sd[tid] += sd[tid + off]; }
    __syncthreads();
  }
  if (tid == 0) { part[b] = sn[0]; part[B_SZ + b] = sd[0]; }
}

__global__ __launch_bounds__(256)
void loss_final_kernel(const float* __restrict__ part, float* __restrict__ out)
{
  __shared__ float sn[256], sd[256];
  int t = threadIdx.x;
  sn[t] = part[t]; sd[t] = part[B_SZ + t]; __syncthreads();
  for (int off = 128; off; off >>= 1) {
    if (t < off) { sn[t] += sn[t + off]; sd[t] += sd[t + off]; }
    __syncthreads();
  }
  if (t == 0) out[0] = sn[0] / sd[0];
}

// ---------------- host orchestration ----------------
extern "C" void kernel_launch(void* const* d_in, const int* in_sizes, int n_in,
                              void* d_out, int out_size, void* d_ws, size_t ws_size,
                              hipStream_t stream)
{
  (void)in_sizes; (void)n_in; (void)out_size; (void)ws_size;
  const float* imgs = (const float*)d_in[0];
  const float* Wqkv = (const float*)d_in[1];
  const float* bqkv = (const float*)d_in[2];
  const float* Wo   = (const float*)d_in[3];
  const float* bo   = (const float*)d_in[4];
  const float* ln1g = (const float*)d_in[5];
  const float* ln1b = (const float*)d_in[6];
  const float* W1   = (const float*)d_in[7];
  const float* b1   = (const float*)d_in[8];
  const float* W2   = (const float*)d_in[9];
  const float* b2   = (const float*)d_in[10];
  const float* ln2g = (const float*)d_in[11];
  const float* ln2b = (const float*)d_in[12];
  const float* mtok = (const float*)d_in[13];
  const float* pW1  = (const float*)d_in[14];
  const float* pb1  = (const float*)d_in[15];
  const float* pW2  = (const float*)d_in[16];
  const float* pb2  = (const float*)d_in[17];
  const float* randK= (const float*)d_in[18];
  const int*   epoch= (const int*)d_in[19];

  // workspace carving (256B-aligned regions)
  char* base = (char*)d_ws;
  size_t off = 0;
  auto alloc = [&](size_t bytes) -> void* {
    void* p = base + off;
    off = (off + bytes + 255) & ~(size_t)255;
    return p;
  };
  float* pos   = (float*)alloc(sizeof(float) * NN * DD);
  float* bufX  = (float*)alloc(sizeof(float) * (size_t)BN * DD);
  float* qkv   = (float*)alloc(sizeof(float) * (size_t)BN * 300);
  float* proj  = (float*)alloc(sizeof(float) * (size_t)BN * DD);
  float* bufY  = (float*)alloc(sizeof(float) * (size_t)BN * DD);
  float* bufT  = (float*)alloc(sizeof(float) * (size_t)BN * DD);
  float* pred  = (float*)alloc(sizeof(float) * (size_t)BN * DD);
  float* sub   = (float*)alloc(sizeof(float) * (size_t)B_SZ * KNET * DD);
  float* cost  = (float*)alloc(sizeof(float) * (size_t)B_SZ * NN * KNET);
  float* Qp    = (float*)alloc(sizeof(float) * (size_t)B_SZ * NN * KNET);
  float* maskb = (float*)alloc(sizeof(float) * (size_t)B_SZ * NN);
  float* part  = (float*)alloc(sizeof(float) * 2 * B_SZ);
  _Float16* Xh     = (_Float16*)alloc(sizeof(_Float16) * (size_t)BN * DPAD);
  _Float16* attOh  = (_Float16*)alloc(sizeof(_Float16) * (size_t)BN * DPAD);
  _Float16* Yh     = (_Float16*)alloc(sizeof(_Float16) * (size_t)BN * DPAD);
  _Float16* hbufH  = (_Float16*)alloc(sizeof(_Float16) * (size_t)CHUNK * FFD);
  _Float16* WqkvT  = (_Float16*)alloc(sizeof(_Float16) * 2 * 320 * 128);
  _Float16* WoT    = (_Float16*)alloc(sizeof(_Float16) * 2 * 128 * 128);
  _Float16* W1T    = (_Float16*)alloc(sizeof(_Float16) * 2 * (size_t)FFD * 128);
  _Float16* W2T    = (_Float16*)alloc(sizeof(_Float16) * 2 * 128 * (size_t)FFD);
  _Float16* pW1T   = (_Float16*)alloc(sizeof(_Float16) * (size_t)PHD * 128);
  _Float16* pW2T   = (_Float16*)alloc(sizeof(_Float16) * 128 * (size_t)PHD);

  auto gemm = [&](const _Float16* Ah, const _Float16* Bt, const float* bias,
                  float* C, _Float16* Ch, int M, int N, int Npad, int Kpad, int act) {
    int tiles = (M >> 5) * (Npad >> 5);
    gemm_kernel<<<(tiles + 7) / 8, 256, 0, stream>>>(Ah, Bt, bias, C, Ch,
                                                     M, N, Npad, Kpad, act);
  };
  auto wt_pack = [&](const float* W, _Float16* Bt, int K, int N, int Kpad, int Npad) {
    int cnt = Npad * Kpad;
    wt_pack_kernel<<<(cnt + 255) / 256, 256, 0, stream>>>(W, Bt, K, N, Kpad, Npad);
  };

  // ---- one-time per-launch setup ----
  pos_kernel<<<(NN * DD + 255) / 256, 256, 0, stream>>>(pos);
  zero_h_kernel<<<((BN * DPAD) + 255) / 256, 256, 0, stream>>>(attOh, (size_t)BN * DPAD);
  for (int l = 0; l < 2; ++l) {
    wt_pack(Wqkv + (size_t)l * DD * 300, WqkvT + (size_t)l * 320 * 128, DD, 300, 128, 320);
    wt_pack(Wo   + (size_t)l * DD * DD,  WoT   + (size_t)l * 128 * 128, DD, DD, 128, 128);
    wt_pack(W1   + (size_t)l * DD * FFD, W1T   + (size_t)l * FFD * 128, DD, FFD, 128, FFD);
    wt_pack(W2   + (size_t)l * FFD * DD, W2T   + (size_t)l * 128 * FFD, FFD, DD, FFD, 128);
  }
  wt_pack(pW1, pW1T, DD, PHD, 128, PHD);
  wt_pack(pW2, pW2T, PHD, DD, PHD, 128);
  a_pack_kernel<<<((BN * DPAD) + 255) / 256, 256, 0, stream>>>(imgs, Xh, BN, DD, DPAD);

  auto layer = [&](const float* xin, float* xout, int l) {
    gemm(Xh, WqkvT + (size_t)l * 320 * 128, bqkv + l * 300, qkv, nullptr, BN, 300, 320, 128, 0);
    attn_kernel<<<B_SZ * NHEADC, 128, 0, stream>>>(qkv, attOh);
    gemm(attOh, WoT + (size_t)l * 128 * 128, bo + l * DD, proj, nullptr, BN, DD, 128, 128, 0);
    add_ln_kernel<<<(BN + 7) / 8, 256, 0, stream>>>(xin, proj, ln1g + l * DD, ln1b + l * DD,
                                                    bufY, Yh, BN);
    for (int c = 0; c < BN; c += CHUNK) {
      gemm(Yh + (size_t)c * DPAD, W1T + (size_t)l * FFD * 128, b1 + l * FFD,
           nullptr, hbufH, CHUNK, FFD, FFD, 128, 1);
      gemm(hbufH, W2T + (size_t)l * 128 * FFD, b2 + l * DD,
           bufT + (size_t)c * DD, nullptr, CHUNK, DD, 128, FFD, 0);
    }
    add_ln_kernel<<<(BN + 7) / 8, 256, 0, stream>>>(bufY, bufT, ln2g + l * DD, ln2b + l * DD,
                                                    xout, Xh, BN);
  };

  // encoder pass 1 (stop-grad features)
  layer(imgs, bufX, 0);
  layer(bufX, bufX, 1);          // bufX = feat (f32), Xh = feat (f16)

  subnet_kernel<<<B_SZ, 256, 0, stream>>>(bufX, sub);
  cost_kernel<<<B_SZ, 256, 0, stream>>>(bufX, sub, cost);
  sinkhorn_kernel<<<B_SZ, 128, 0, stream>>>(cost, epoch, Qp);
  mask_kernel<<<B_SZ, 128, 0, stream>>>(Qp, randK, epoch, maskb);
  build_x_kernel<<<((BN * DPAD) + 255) / 256, 256, 0, stream>>>(imgs, mtok, maskb, pos,
                                                                bufX, Xh);

  // encoder pass 2 (masked input)
  layer(bufX, bufX, 0);
  layer(bufX, bufX, 1);

  // predictor (hidden chunk reuses hbufH, stride PHD)
  for (int c = 0; c < BN; c += CHUNK) {
    gemm(Xh + (size_t)c * DPAD, pW1T, pb1, nullptr, hbufH, CHUNK, PHD, PHD, 128, 2);
    gemm(hbufH, pW2T, pb2, pred + (size_t)c * DD, nullptr, CHUNK, DD, 128, PHD, 0);
  }

  loss_partial_kernel<<<B_SZ, 128, 0, stream>>>(pred, imgs, maskb, part);
  loss_final_kernel<<<1, 256, 0, stream>>>(part, (float*)d_out);
}
